// WaveletScatteringTransform_34059090657746
// MI455X (gfx1250) — matrix operand order repaired and checked by reading, not compiled, not run on hardware
//
#include <hip/hip_runtime.h>
#include <hip/hip_bf16.h>
#include <math.h>

// ---------------------------------------------------------------------------
// Wavelet scattering on MI455X (gfx1250, wave32, WMMA).
// T = 16000 = 128*125 -> every FFT/IFFT is two dense complex GEMMs
// (125x125x128 and 125x128x128) + an elementwise twiddle, executed with
// v_wmma_f32_16x16x32_f16 (f16 operands, f32 accumulate).
//   - signal is ALWAYS the WMMA A operand (row-major in LDS, contiguous)
//   - DFT matrices are symmetric -> B fragments load as contiguous rows
//   - stage outputs stored transposed (one ds_store_b128) when the next
//     stage needs the flipped layout:
//     time[n1][n2] -fwd1,T-> [k2][n1] -fwd2-> spec[k2][k1]
//                  -inv1,T-> [n1][k2] -inv2-> time[n1][n2]
//   - forward stage 1 has a REAL A operand (windowed x, |ifft|): 2 WMMAs/k-step
//   - complex stages use Karatsuba (3 WMMAs/k-step); f16 adds co-execute
//   - LDS row pitch 136 halves (272B = 68 dwords) -> row-to-row bank shift 4,
//     conflict-free ds_load_b128 across the 16 row-lanes
// Workspace requirement: ~51 MB.
// ---------------------------------------------------------------------------

typedef _Float16 v16h __attribute__((ext_vector_type(16)));
typedef _Float16 h8   __attribute__((ext_vector_type(8)));
typedef float    v8f  __attribute__((ext_vector_type(8)));

#define T_LEN 16000
#define NN1   128          // k1 / n1 dimension
#define NN2   125          // k2 / n2 dimension
#define PAD   128
#define PLANE (PAD*PAD)    // 16384 elements per matrix plane (global, pitch 128)
#define LPITCH 136         // LDS plane pitch in halves (bank-conflict padding)
#define LSIZE  (PAD*LPITCH)
#define LIDX(r,c) ((r)*LPITCH+(c))
#define BATCH 8
#define NP1   84
#define NPATH 252
#define NCHAN 337
#define OUT_T 125

// workspace byte offsets (all 64B aligned)
#define OFF_WIN   0u
#define OFF_MATS  65536u            // 4 mats * 2 planes * 16384 f16 = 256 KB
#define OFF_TW    327680u           // 16384 * float2                = 128 KB
#define OFF_E125  458752u           // 125*125 * float2 IDFT table   = 125 KB
#define OFF_PSI1  589824u           // 84*16000 f32
#define OFF_PSI2  5965824u          // 7*16000 f32
#define OFF_PHI   6413824u          // 16000 f32
#define OFF_XSPEC 6477824u          // 8*16000 float2
#define OFF_U1H   7501824u          // 8*84*16000 * (2 x f16) = 43 MB
#define WS_NEED   50509824u

struct __attribute__((aligned(4))) h2 { _Float16 x, y; };

#define EPI_NONE 0
#define EPI_TW   1
#define EPI_ABS  2

// ---------------------------------------------------------------------------
// WMMA helper (CDNA5 16x16x32 f16 -> f32)
// ---------------------------------------------------------------------------
__device__ inline v8f wmma_f16(v16h a, v16h b, v8f c) {
  // 8 args: (neg_a, A, neg_b, B, c_mod, C, reuse_a, reuse_b)
  return __builtin_amdgcn_wmma_f32_16x16x32_f16(false, a, false, b, (short)0, c,
                                                false, false);
}

// A fragment: 16x32 tile of a row-major plane (ISA 7.12.2 16-bit A layout).
// lane L: M = L&15 ; element pairs cover K = k0 + (L>=16?8:0) + {0..7,16..23}.
__device__ inline v16h frag_a(const _Float16* base, int pitch, int m0, int k0,
                              int lane) {
  const _Float16* p = base + (m0 + (lane & 15)) * pitch + k0 +
                      ((lane & 16) ? 8 : 0);
  const h8 lo = *(const h8*)p;
  const h8 hi = *(const h8*)(p + 16);
  return __builtin_shufflevector(lo, hi, 0, 1, 2, 3, 4, 5, 6, 7,
                                 8, 9, 10, 11, 12, 13, 14, 15);
}

// B fragment of a SYMMETRIC matrix: element pairs cover K = k0 + (L>=16?8:0) + {0..7,16..23} at
// column N = n0 + (L&15).  B[K][N] = M[K][N] = M[N][K] -> contiguous row read.
__device__ inline v16h frag_b_sym(const _Float16* base, int k0, int n0,
                                  int lane) {
  const _Float16* p = base + (n0 + (lane & 15)) * PAD + k0 +
                      ((lane & 16) ? 8 : 0);
  const h8 lo = *(const h8*)p;
  const h8 hi = *(const h8*)(p + 16);
  return __builtin_shufflevector(lo, hi, 0, 1, 2, 3, 4, 5, 6, 7,
                                 8, 9, 10, 11, 12, 13, 14, 15);
}

// Shared epilogue + store for one 16x16 C tile.
__device__ inline void store_tile(v8f accRe, v8f accIm, _Float16* dRe,
                                  _Float16* dIm, int tm, int tn, int lane,
                                  int epi, bool storeTrans, bool storeIm,
                                  const float* tw, float twSign, float scale) {
  const int n  = tn + (lane & 15);
  const int m0 = tm + ((lane & 16) ? 8 : 0);
  h8 vr, vi;
#pragma unroll
  for (int r = 0; r < 8; ++r) {
    const int m = m0 + r;
    float cr = accRe[r], ci = accIm[r];
    if (epi == EPI_TW) {
      const float twr = tw[(m * PAD + n) * 2];     // depends on m*n: symmetric
      const float twi = twSign * tw[(m * PAD + n) * 2 + 1];
      const float nr = cr * twr - ci * twi;
      const float ni = cr * twi + ci * twr;
      cr = nr * scale;
      ci = ni * scale;
    } else if (epi == EPI_ABS) {
      cr = sqrtf(cr * cr + ci * ci);
      ci = 0.f;
    }
    vr[r] = (_Float16)cr;
    vi[r] = (_Float16)ci;
  }
  if (storeTrans) {                        // P^T: contiguous 16B per plane
    *(h8*)(dRe + n * LPITCH + m0) = vr;
    if (storeIm) *(h8*)(dIm + n * LPITCH + m0) = vi;
  } else {
#pragma unroll
    for (int r = 0; r < 8; ++r) {
      dRe[(m0 + r) * LPITCH + n] = vr[r];
      if (storeIm) dIm[(m0 + r) * LPITCH + n] = vi[r];
    }
  }
}

// ---------------------------------------------------------------------------
// Complex GEMM stage (Karatsuba, 3 WMMAs / k-step):
//   P1 = Ar*Br, P2 = Ai*Bi, P3 = (Ar+Ai)(Br+Bi)
//   Re = P1 - P2 ; Im = P3 - P1 - P2
// The f16 fragment adds dual-issue with the XDL WMMA pipe.
// ---------------------------------------------------------------------------
__device__ inline void cgemm_cplx(const _Float16* aRe, const _Float16* aIm,
                                  const _Float16* mRe, const _Float16* mIm,
                                  _Float16* dRe, _Float16* dIm,
                                  int epi, bool storeTrans, bool storeIm,
                                  const float* tw, float twSign, float scale) {
  const int tid  = threadIdx.x;
  const int lane = tid & 31;
  const int wave = tid >> 5;
  for (int t = wave; t < 64; t += 8) {     // 8x8 grid of 16x16 output tiles
    const int tm = (t >> 3) << 4;
    const int tn = (t & 7) << 4;
    v8f p1 = {}, p2 = {}, p3 = {};
#pragma unroll
    for (int k0 = 0; k0 < PAD; k0 += 32) {
      const v16h ar = frag_a(aRe, LPITCH, tm, k0, lane);
      const v16h ai = frag_a(aIm, LPITCH, tm, k0, lane);
      const v16h br = frag_b_sym(mRe, k0, tn, lane);
      const v16h bi = frag_b_sym(mIm, k0, tn, lane);
      const v16h as = ar + ai;
      const v16h bs = br + bi;
      p1 = wmma_f16(ar, br, p1);
      p2 = wmma_f16(ai, bi, p2);
      p3 = wmma_f16(as, bs, p3);
    }
    v8f accRe, accIm;
#pragma unroll
    for (int r = 0; r < 8; ++r) {
      accRe[r] = p1[r] - p2[r];
      accIm[r] = p3[r] - p1[r] - p2[r];
    }
    store_tile(accRe, accIm, dRe, dIm, tm, tn, lane, epi, storeTrans, storeIm,
               tw, twSign, scale);
  }
}

// Real-A GEMM stage (2 WMMAs / k-step): A imag == 0.
__device__ inline void cgemm_real(const _Float16* aRe,
                                  const _Float16* mRe, const _Float16* mIm,
                                  _Float16* dRe, _Float16* dIm,
                                  const float* tw, float twSign, float scale) {
  const int tid  = threadIdx.x;
  const int lane = tid & 31;
  const int wave = tid >> 5;
  for (int t = wave; t < 64; t += 8) {
    const int tm = (t >> 3) << 4;
    const int tn = (t & 7) << 4;
    v8f accRe = {}, accIm = {};
#pragma unroll
    for (int k0 = 0; k0 < PAD; k0 += 32) {
      const v16h ar = frag_a(aRe, LPITCH, tm, k0, lane);
      const v16h br = frag_b_sym(mRe, k0, tn, lane);
      const v16h bi = frag_b_sym(mIm, k0, tn, lane);
      accRe = wmma_f16(ar, br, accRe);
      accIm = wmma_f16(ar, bi, accIm);
    }
    store_tile(accRe, accIm, dRe, dIm, tm, tn, lane, EPI_TW, true, true,
               tw, twSign, scale);
  }
}

// ---------------------------------------------------------------------------
// Full length-16000 transforms, in-LDS ping-pong.
// ---------------------------------------------------------------------------
__device__ inline void fwd_fft(_Float16* sRe, _Float16* sIm,
                               _Float16* dRe, _Float16* dIm,
                               const _Float16* mats, const float* tw) {
  const _Float16* mf125 = mats + 0 * 2 * PLANE;
  const _Float16* mf128 = mats + 2 * 2 * PLANE;
  __syncthreads();
  // REAL input: P[n1][k2] = sum_n2 time[n1][n2]*F125[n2][k2]; tw; ^T -> [k2][n1]
  cgemm_real(sRe, mf125, mf125 + PLANE, dRe, dIm, tw, 1.f, 1.f);
  __syncthreads();
  // spec[k2][k1] = sum_n1 D[k2][n1] * F128[n1][k1]
  cgemm_cplx(dRe, dIm, mf128, mf128 + PLANE, sRe, sIm,
             EPI_NONE, false, true, tw, 0.f, 1.f);
  __syncthreads();
}

__device__ inline void inv_fft_abs(_Float16* sRe, _Float16* sIm,
                                   _Float16* dRe, _Float16* dIm,
                                   const _Float16* mats, const float* tw) {
  const _Float16* mi125 = mats + 1 * 2 * PLANE;
  const _Float16* mi128 = mats + 3 * 2 * PLANE;
  __syncthreads();
  // P[k2][n1] = sum_k1 spec[k2][k1]*G128[k1][n1]; conj tw; 1/N; ^T -> [n1][k2]
  cgemm_cplx(sRe, sIm, mi128, mi128 + PLANE, dRe, dIm,
             EPI_TW, true, true, tw, -1.f, 1.f / 16000.f);
  __syncthreads();
  // time^T[n1][n2] = sum_k2 D[n1][k2]*G125[k2][n2]; |.| -> real time[n1][n2]
  cgemm_cplx(dRe, dIm, mi125, mi125 + PLANE, sRe, sIm,
             EPI_ABS, false, false, tw, 0.f, 1.f);
  __syncthreads();
}

// ---------------------------------------------------------------------------
// Lowpass (phi) + downsample-by-128 + eval-mode log-normalization.
// fold[k2] = sum_k1 spec[k2][k1]*phi ; S[m] = Re IDFT125(fold) / N (table).
// ---------------------------------------------------------------------------
__device__ inline void lowpass_out(const _Float16* sRe, const _Float16* sIm,
                                   const float* __restrict__ phip,
                                   const float* __restrict__ e125,
                                   float* __restrict__ outp,
                                   float* foldRe, float* foldIm) {
  const int tid = threadIdx.x;
  if (tid < NN2) {
    float fr = 0.f, fi = 0.f;
    for (int k1 = 0; k1 < PAD; k1 += 8) {
      const h8 re = *(const h8*)(sRe + tid * LPITCH + k1);
      const h8 im = *(const h8*)(sIm + tid * LPITCH + k1);
      const float4 p0 = *(const float4*)(phip + tid * PAD + k1);
      const float4 p1 = *(const float4*)(phip + tid * PAD + k1 + 4);
      fr += (float)re[0] * p0.x + (float)re[1] * p0.y + (float)re[2] * p0.z +
            (float)re[3] * p0.w + (float)re[4] * p1.x + (float)re[5] * p1.y +
            (float)re[6] * p1.z + (float)re[7] * p1.w;
      fi += (float)im[0] * p0.x + (float)im[1] * p0.y + (float)im[2] * p0.z +
            (float)im[3] * p0.w + (float)im[4] * p1.x + (float)im[5] * p1.y +
            (float)im[6] * p1.z + (float)im[7] * p1.w;
    }
    foldRe[tid] = fr;
    foldIm[tid] = fi;
  }
  __syncthreads();
  if (tid < OUT_T) {
    float acc = 0.f;
    for (int k2 = 0; k2 < NN2; ++k2) {
      const float c = e125[(k2 * NN2 + tid) * 2];
      const float s = e125[(k2 * NN2 + tid) * 2 + 1];
      acc += foldRe[k2] * c - foldIm[k2] * s;
    }
    acc *= (1.f / 16000.f);
    const float Sn = acc / (1.f + 1e-8f);
    const float sg = (Sn > 0.f) ? 1.f : ((Sn < 0.f) ? -1.f : 0.f);
    const float lg = sg * log1pf(fabsf(Sn) * 1e5f) * 1e-5f;
    outp[tid] = 0.7f * Sn + 0.3f * lg;
  }
  __syncthreads();
}

// ---------------------------------------------------------------------------
// Table builder.  Exact phase reduction: theta = 2*pi*((r*c) mod S)/S.
// ---------------------------------------------------------------------------
__global__ __launch_bounds__(256) void k_build(float* __restrict__ win,
                                               _Float16* __restrict__ mats,
                                               float* __restrict__ tw,
                                               float* __restrict__ e125,
                                               float* __restrict__ psi1,
                                               float* __restrict__ psi2,
                                               float* __restrict__ phi) {
  const float TWO_PI = 6.28318530717958647692f;
  const int gid = blockIdx.x * blockDim.x + threadIdx.x;
  const int gsz = gridDim.x * blockDim.x;

  for (int i = gid; i < T_LEN; i += gsz) {
    win[i] = 0.92f * 0.5f * (1.f - cosf(TWO_PI * (float)i / 16000.f)) + 0.08f;
    // permuted spectral index: i = k2*128 + k1  ->  k = 125*k1 + k2
    const int k = 125 * (i & 127) + (i >> 7);
    const float om = (float)k / 16000.f;
    const float sp = 0.4f / 128.f;
    phi[i] = expf(-(om * om) / (2.f * sp * sp)) +
             expf(-((om - 1.f) * (om - 1.f)) / (2.f * sp * sp));
  }
  for (int i = gid; i < 4 * PLANE; i += gsz) {
    const int id = i >> 14;          // 0:F125 fwd 1:F125 inv 2:F128 fwd 3:F128 inv
    const int rc = i & (PLANE - 1);
    const int r = rc >> 7, c = rc & 127;
    const int S = (id < 2) ? NN2 : NN1;
    float re = 0.f, im = 0.f;
    if (r < S && c < S) {
      const int ph = (r * c) % S;
      const float th = TWO_PI * (float)ph / (float)S;
      re = cosf(th);
      im = (id & 1) ? sinf(th) : -sinf(th);
    }
    const int base = id * 2 * PLANE;
    mats[base + rc]         = (_Float16)re;
    mats[base + PLANE + rc] = (_Float16)im;
  }
  for (int i = gid; i < PLANE; i += gsz) {
    const int r = i >> 7, c = i & 127;
    const int ph = (r * c) % T_LEN;
    const float th = TWO_PI * (float)ph / 16000.f;
    tw[2 * i]     = cosf(th);
    tw[2 * i + 1] = -sinf(th);
  }
  for (int i = gid; i < NN2 * NN2; i += gsz) {
    const int r = i / NN2, c = i % NN2;
    const float th = TWO_PI * (float)((r * c) % NN2) / (float)NN2;
    e125[2 * i]     = cosf(th);
    e125[2 * i + 1] = sinf(th);
  }
  for (int i = gid; i < NP1 * T_LEN; i += gsz) {
    const int p = i / T_LEN, j = i % T_LEN;
    const int k = 125 * (j & 127) + (j >> 7);
    const float om = (float)k / 16000.f;
    const float xi = 0.5f * exp2f(-(float)(p + 1) / 12.f);
    const float sg = xi * (exp2f(1.f / 12.f) - 1.f) * 2.f;
    psi1[i] = expf(-((om - xi) * (om - xi)) / (2.f * sg * sg));
  }
  for (int i = gid; i < 7 * T_LEN; i += gsz) {
    const int p = i / T_LEN, j = i % T_LEN;
    const int k = 125 * (j & 127) + (j >> 7);
    const float om = (float)k / 16000.f;
    const float xi = 0.5f * exp2f(-(float)(p + 1));
    const float sg = 0.5f * xi;
    psi2[i] = expf(-((om - xi) * (om - xi)) / (2.f * sg * sg));
  }
}

// ---------------------------------------------------------------------------
// Stage kernels: each block = one transform chain, signal resident in LDS.
// ---------------------------------------------------------------------------
__global__ __launch_bounds__(256) void k_fft_x(const float* __restrict__ x,
                                               const float* __restrict__ win,
                                               const _Float16* __restrict__ mats,
                                               const float* __restrict__ tw,
                                               const float* __restrict__ phip,
                                               const float* __restrict__ e125,
                                               float2* __restrict__ xspec,
                                               float* __restrict__ out) {
  __shared__ alignas(16) _Float16 sRe[LSIZE], sIm[LSIZE], dRe[LSIZE], dIm[LSIZE];
  __shared__ float foldRe[NN2], foldIm[NN2];
  const int b = blockIdx.x;
  const int tid = threadIdx.x;
  // time layout [n1][n2]: LDS = x[n1 + 128*n2]*win, zero-padded n2 >= 125
  for (int i = tid; i < PLANE; i += 256) {
    const int n1 = i >> 7, n2 = i & 127;
    float v = 0.f;
    if (n2 < NN2) {
      const int nsrc = n1 + 128 * n2;
      v = x[b * T_LEN + nsrc] * win[nsrc];
    }
    sRe[LIDX(n1, n2)] = (_Float16)v;       // real input: sIm not needed
  }
  fwd_fft(sRe, sIm, dRe, dIm, mats, tw);   // s <- spec [k2][k1]
  for (int i = tid; i < T_LEN; i += 256)
    xspec[b * T_LEN + i] =
        make_float2((float)sRe[LIDX(i >> 7, i & 127)],
                    (float)sIm[LIDX(i >> 7, i & 127)]);
  lowpass_out(sRe, sIm, phip, e125, out + (b * NCHAN + 0) * OUT_T, foldRe, foldIm);
}

__global__ __launch_bounds__(256) void k_order1(const float2* __restrict__ xspec,
                                                const float* __restrict__ psi1p,
                                                const float* __restrict__ phip,
                                                const float* __restrict__ e125,
                                                const _Float16* __restrict__ mats,
                                                const float* __restrict__ tw,
                                                h2* __restrict__ u1h,
                                                float* __restrict__ out) {
  __shared__ alignas(16) _Float16 sRe[LSIZE], sIm[LSIZE], dRe[LSIZE], dIm[LSIZE];
  __shared__ float foldRe[NN2], foldIm[NN2];
  const int blk = blockIdx.x;
  const int b = blk / NP1, p1 = blk % NP1;
  const int tid = threadIdx.x;
  for (int i = tid; i < PLANE; i += 256) {   // spec [k2][k1] * psi1, zero pad
    float re = 0.f, im = 0.f;
    if (i < T_LEN) {
      const float2 X = xspec[b * T_LEN + i];
      const float f = psi1p[p1 * T_LEN + i];
      re = X.x * f;
      im = X.y * f;
    }
    sRe[LIDX(i >> 7, i & 127)] = (_Float16)re;
    sIm[LIDX(i >> 7, i & 127)] = (_Float16)im;
  }
  inv_fft_abs(sRe, sIm, dRe, dIm, mats, tw); // s <- |ifft(X*psi1)| time [n1][n2]
  fwd_fft(sRe, sIm, dRe, dIm, mats, tw);     // s <- U1h spectrum [k2][k1]
  for (int i = tid; i < T_LEN; i += 256) {
    h2 v;
    v.x = sRe[LIDX(i >> 7, i & 127)];
    v.y = sIm[LIDX(i >> 7, i & 127)];
    u1h[(b * NP1 + p1) * T_LEN + i] = v;
  }
  lowpass_out(sRe, sIm, phip, e125, out + (b * NCHAN + 1 + p1) * OUT_T,
              foldRe, foldIm);
}

__global__ __launch_bounds__(256) void k_order2(const h2* __restrict__ u1h,
                                                const float* __restrict__ psi2p,
                                                const float* __restrict__ phip,
                                                const float* __restrict__ e125,
                                                const _Float16* __restrict__ mats,
                                                const float* __restrict__ tw,
                                                float* __restrict__ out) {
  __shared__ alignas(16) _Float16 sRe[LSIZE], sIm[LSIZE], dRe[LSIZE], dIm[LSIZE];
  __shared__ float foldRe[NN2], foldIm[NN2];
  const int blk = blockIdx.x;
  const int b = blk / NPATH, p = blk % NPATH;
  // decode path p -> (i1, i2): row-major over j1(i1) < j2 pairs
  int i1 = 0, i2 = 0, cnt = 0;
  for (int a = 0; a < NP1; ++a) {
    const int j1 = a / 12;
    const int c = 6 - j1;
    if (p < cnt + c) { i1 = a; i2 = (j1 + 1) + (p - cnt); break; }
    cnt += c;
  }
  const int tid = threadIdx.x;
  for (int i = tid; i < PLANE; i += 256) {
    float re = 0.f, im = 0.f;
    if (i < T_LEN) {
      const h2 v = u1h[(b * NP1 + i1) * T_LEN + i];
      const float f = psi2p[i2 * T_LEN + i];
      re = (float)v.x * f;
      im = (float)v.y * f;
    }
    sRe[LIDX(i >> 7, i & 127)] = (_Float16)re;
    sIm[LIDX(i >> 7, i & 127)] = (_Float16)im;
  }
  inv_fft_abs(sRe, sIm, dRe, dIm, mats, tw); // s <- |ifft(U1h*psi2)|
  fwd_fft(sRe, sIm, dRe, dIm, mats, tw);     // s <- U2h spectrum
  lowpass_out(sRe, sIm, phip, e125, out + (b * NCHAN + 85 + p) * OUT_T,
              foldRe, foldIm);
}

// ---------------------------------------------------------------------------
extern "C" void kernel_launch(void* const* d_in, const int* in_sizes, int n_in,
                              void* d_out, int out_size, void* d_ws, size_t ws_size,
                              hipStream_t stream) {
  (void)in_sizes; (void)n_in; (void)out_size; (void)ws_size;
  const float* x = (const float*)d_in[0];
  float* out = (float*)d_out;
  char* ws = (char*)d_ws;

  float*    win   = (float*)   (ws + OFF_WIN);
  _Float16* mats  = (_Float16*)(ws + OFF_MATS);
  float*    tw    = (float*)   (ws + OFF_TW);
  float*    e125  = (float*)   (ws + OFF_E125);
  float*    psi1  = (float*)   (ws + OFF_PSI1);
  float*    psi2  = (float*)   (ws + OFF_PSI2);
  float*    phi   = (float*)   (ws + OFF_PHI);
  float2*   xspec = (float2*)  (ws + OFF_XSPEC);
  h2*       u1h   = (h2*)      (ws + OFF_U1H);

  k_build <<<1024, 256, 0, stream>>>(win, mats, tw, e125, psi1, psi2, phi);
  k_fft_x <<<BATCH, 256, 0, stream>>>(x, win, mats, tw, phi, e125, xspec, out);
  k_order1<<<BATCH * NP1, 256, 0, stream>>>(xspec, psi1, phi, e125, mats, tw,
                                            u1h, out);
  k_order2<<<BATCH * NPATH, 256, 0, stream>>>(u1h, psi2, phi, e125, mats, tw,
                                              out);
}
